// LinearCatVAE_22754736734572
// MI455X (gfx1250) — matrix operand order repaired and checked by reading, not compiled, not run on hardware
//
#include <hip/hip_runtime.h>
#include <math.h>

typedef __attribute__((ext_vector_type(16))) _Float16 v16h;
typedef __attribute__((ext_vector_type(8)))  float    v8f;

union HFrag { v16h v; _Float16 h[16]; };

#define LOG2PI 1.8378770664093453f

// ---------------------------------------------------------------------------
// WMMA fragment loaders (ISA 7.12.2 16-bit layouts), f32->f16 on the fly.
// A 16x32 (MxK): lane L<16: M=L, VGPR v holds K pair; lanes 16-31: same M, K+8.
// B 32x16 (KxN): lane L<16: N=L, VGPR v holds K pair (same K as A); lanes 16-31: same N, K+8.
// Fast variants: no bounds checks (steady-state K loop).
// Guarded variants: clamp index + unconditional load + select (no exec branches).
// ---------------------------------------------------------------------------
__device__ __forceinline__ v16h ld_afrag_f(const float* p, int ldm, int ldk,
                                           int M0, int kb, int lane) {
  HFrag f;
  int hi = lane >> 4;
  int ro = (M0 + (lane & 15)) * ldm;
#pragma unroll
  for (int v = 0; v < 8; ++v) {
    int k0 = kb + ((v & 4) ? 16 : 0) + 2 * (v & 3) + 8 * hi;
    f.h[2 * v]     = (_Float16)p[ro + k0 * ldk];
    f.h[2 * v + 1] = (_Float16)p[ro + (k0 + 1) * ldk];
  }
  return f.v;
}

__device__ __forceinline__ v16h ld_afrag_g(const float* p, int ldm, int ldk,
                                           int M0, int kb, int kmax, int lane) {
  HFrag f;
  int hi = lane >> 4;
  int ro = (M0 + (lane & 15)) * ldm;
#pragma unroll
  for (int v = 0; v < 8; ++v) {
    int k0 = kb + ((v & 4) ? 16 : 0) + 2 * (v & 3) + 8 * hi;
    float a0 = p[ro + min(k0, kmax - 1) * ldk];
    float a1 = p[ro + min(k0 + 1, kmax - 1) * ldk];
    f.h[2 * v]     = (_Float16)((k0     < kmax) ? a0 : 0.f);
    f.h[2 * v + 1] = (_Float16)((k0 + 1 < kmax) ? a1 : 0.f);
  }
  return f.v;
}

__device__ __forceinline__ v16h ld_bfrag_f(const float* p, int ldk, int ldn,
                                           int N0, int kb, int lane) {
  HFrag f;
  int hi = lane >> 4;
  int no = (N0 + (lane & 15)) * ldn;
#pragma unroll
  for (int v = 0; v < 8; ++v) {
    int k0 = kb + ((v & 4) ? 16 : 0) + 2 * (v & 3) + 8 * hi;
    f.h[2 * v]     = (_Float16)p[no + k0 * ldk];
    f.h[2 * v + 1] = (_Float16)p[no + (k0 + 1) * ldk];
  }
  return f.v;
}

__device__ __forceinline__ v16h ld_bfrag_g(const float* p, int ldk, int ldn,
                                           int N0, int kb, int kmax, int nmax,
                                           int lane) {
  HFrag f;
  int hi = lane >> 4;
  int n  = N0 + (lane & 15);
  int no = min(n, nmax - 1) * ldn;
  bool nok = (n < nmax);
#pragma unroll
  for (int v = 0; v < 8; ++v) {
    int k0 = kb + ((v & 4) ? 16 : 0) + 2 * (v & 3) + 8 * hi;
    float b0 = p[no + min(k0, kmax - 1) * ldk];
    float b1 = p[no + min(k0 + 1, kmax - 1) * ldk];
    f.h[2 * v]     = (_Float16)((nok && k0     < kmax) ? b0 : 0.f);
    f.h[2 * v + 1] = (_Float16)((nok && k0 + 1 < kmax) ? b1 : 0.f);
  }
  return f.v;
}

__device__ __forceinline__ float block_reduce_sum(float val, float* sm) {
  int t = threadIdx.x;
  sm[t] = val; __syncthreads();
  for (int s = blockDim.x >> 1; s; s >>= 1) {
    if (t < s) sm[t] += sm[t + s];
    __syncthreads();
  }
  float r = sm[0]; __syncthreads();
  return r;
}

// ---- small scratch zero + float-max key init ----
__global__ void vae_init_small(float* p, int* mxi, int n) {
  int i = blockIdx.x * blockDim.x + threadIdx.x;
  if (i < n) p[i] = 0.f;
  if (i < 128) mxi[i] = (int)0x80000000;
}

// ---- lx = log(x+1); logits = 0 ----
__global__ void vae_precompute(const float* x, float* lx, float* logits, int n) {
  int i = blockIdx.x * blockDim.x + threadIdx.x;
  if (i < n) { lx[i] = logf(x[i] + 1.f); logits[i] = 0.f; }
}

// ---- CSR row starts from sorted psi_rows ----
__global__ void vae_rowstart(const int* rows, int* rs, int nnz, int dm1) {
  int e = blockIdx.x * blockDim.x + threadIdx.x;
  if (e == 0) { rs[rows[0]] = 0; rs[dm1] = nnz; }
  else if (e < nnz && rows[e] != rows[e - 1]) rs[rows[e]] = e;
}

// ---- hx[b][r] = sum vals * lx[b][cols]; one wave per (row, batch) ----
__global__ void vae_hx(const float* lx, const int* rs, const int* cols, const float* vals,
                       float* hx, int D, int dm1) {
  int r    = blockIdx.x;
  int wave = threadIdx.x >> 5;
  int lane = threadIdx.x & 31;
  int b    = blockIdx.y * (blockDim.x >> 5) + wave;
  int s = rs[r], e = rs[r + 1];
  float acc = 0.f;
  for (int j = s + lane; j < e; j += 32)
    acc += vals[j] * lx[b * D + cols[j]];
  for (int m = 16; m; m >>= 1) acc += __shfl_xor(acc, m, 32);
  if (lane == 0) hx[b * dm1 + r] = acc;
}

// ---- generic split-K WMMA GEMM onto a 128x128 f32 output (atomic accumulate) ----
__global__ void vae_gemm128(const float* A, int a_ldm, int a_ldk,
                            const float* B, int b_ldk, int b_ldn,
                            float* C, int kmax, int kch) {
  int lane = threadIdx.x & 31;
  int mt = (blockIdx.x >> 3) * 16, nt = (blockIdx.x & 7) * 16;
  int kb  = blockIdx.y * kch * 32;
  int kb1 = min(kb + kch * 32, kmax);
  v8f acc = {};
  // steady state: no bounds checks
  for (; kb + 32 <= kb1; kb += 32) {
    v16h a = ld_afrag_f(A, a_ldm, a_ldk, mt, kb, lane);
    v16h b = ld_bfrag_f(B, b_ldk, b_ldn, nt, kb, lane);
    acc = __builtin_amdgcn_wmma_f32_16x16x32_f16(false, a, false, b, (short)0, acc,
                                                 false, false);
  }
  // tail (at most one partial 32-step, uniform branch -> EXEC all-1 at WMMA)
  if (kb < kb1) {
    v16h a = ld_afrag_g(A, a_ldm, a_ldk, mt, kb, kb1, lane);
    v16h b = ld_bfrag_g(B, b_ldk, b_ldn, nt, kb, kb1, 128, lane);
    acc = __builtin_amdgcn_wmma_f32_16x16x32_f16(false, a, false, b, (short)0, acc,
                                                 false, false);
  }
  int hi = lane >> 4;
  int n  = nt + (lane & 15);
#pragma unroll
  for (int v = 0; v < 8; ++v) {
    int m = mt + v + 8 * hi;
    atomicAdd(&C[m * 128 + n], acc[v]);
  }
}

// ---- mu = z_mean @ dec_w^T (WMMA, K=128); diff = eta - mu; diff2[b] += sum diff^2 ----
__global__ void vae_mu_diff(const float* zmean, const float* dec_w, const float* eta,
                            float* diff, float* diff2, int dm1) {
  int lane = threadIdx.x & 31;
  int nt = blockIdx.x * 16;   // column tile (dm1 dim)
  int mt = blockIdx.y * 16;   // batch tile
  v8f acc = {};
  if (nt + 16 <= dm1) {       // uniform branch per wave
#pragma unroll
    for (int kb = 0; kb < 128; kb += 32) {
      v16h a = ld_afrag_f(zmean, 128, 1, mt, kb, lane);
      v16h b = ld_bfrag_f(dec_w, 1, 128, nt, kb, lane);  // B(k=h,n=j)=dec_w[j*128+h]
      acc = __builtin_amdgcn_wmma_f32_16x16x32_f16(false, a, false, b, (short)0, acc,
                                                   false, false);
    }
  } else {
#pragma unroll
    for (int kb = 0; kb < 128; kb += 32) {
      v16h a = ld_afrag_f(zmean, 128, 1, mt, kb, lane);
      v16h b = ld_bfrag_g(dec_w, 1, 128, nt, kb, 128, dm1, lane);
      acc = __builtin_amdgcn_wmma_f32_16x16x32_f16(false, a, false, b, (short)0, acc,
                                                   false, false);
    }
  }
  int hi  = lane >> 4;
  int col = nt + (lane & 15);
#pragma unroll
  for (int v = 0; v < 8; ++v) {
    int row = mt + v + 8 * hi;
    float d = 0.f;
    if (col < dm1) {
      d = eta[row * dm1 + col] - acc[v];
      diff[row * dm1 + col] = d;
    }
    float s = d * d;
    s += __shfl_xor(s, 1, 32);
    s += __shfl_xor(s, 2, 32);
    s += __shfl_xor(s, 4, 32);
    s += __shfl_xor(s, 8, 32);
    if ((lane & 15) == 0) atomicAdd(&diff2[row], s);
  }
}

// ---- prior: sum of z_mean^2 ----
__global__ void vae_prior(const float* z, float* prior) {
  __shared__ float sm[256];
  float a = 0.f;
  for (int i = blockIdx.x * 256 + threadIdx.x; i < 16384; i += gridDim.x * 256) {
    float v = z[i]; a += v * v;
  }
  float r = block_reduce_sum(a, sm);
  if (threadIdx.x == 0) atomicAdd(prior, r);
}

// ---- scatter logits[b][col] += eta[b][row]*val (coalesced over nnz) ----
__global__ void vae_scatter(const int* rows, const int* cols, const float* vals,
                            const float* eta, float* logits, int D, int dm1, int nnz) {
  int e = blockIdx.x * blockDim.x + threadIdx.x;
  int b = blockIdx.y;
  if (e < nnz) {
    atomicAdd(&logits[b * D + cols[e]], eta[b * dm1 + rows[e]] * vals[e]);
  }
}

// ---- per-batch logits max via int-keyed atomicMax ----
__global__ void vae_rowmax(const float* logits, int* mxi, int D) {
  int b = blockIdx.y;
  __shared__ float sm[256];
  float m = -3.0e38f;
  for (int d = blockIdx.x * 256 + threadIdx.x; d < D; d += gridDim.x * 256)
    m = fmaxf(m, logits[b * D + d]);
  int t = threadIdx.x;
  sm[t] = m; __syncthreads();
  for (int s = 128; s; s >>= 1) {
    if (t < s) sm[t] = fmaxf(sm[t], sm[t + s]);
    __syncthreads();
  }
  if (t == 0) {
    int fi = __float_as_int(sm[0]);
    int key = (fi >= 0) ? fi : (fi ^ 0x7FFFFFFF);
    atomicMax(&mxi[b], key);
  }
}

// ---- per-batch sums: sum exp(l-mx), sum x*l, sum x, sum lgamma(x+1) ----
__global__ void vae_rowsum(const float* logits, const float* x, const int* mxi,
                           float* se, float* sxl, float* sx, float* sgl, int D) {
  int b = blockIdx.y;
  int k = mxi[b];
  float mx = __int_as_float(k >= 0 ? k : (k ^ 0x7FFFFFFF));
  float a_se = 0.f, a_sxl = 0.f, a_sx = 0.f, a_sgl = 0.f;
  for (int d = blockIdx.x * 256 + threadIdx.x; d < D; d += gridDim.x * 256) {
    float lg = logits[b * D + d];
    float xv = x[b * D + d];
    a_se  += expf(lg - mx);
    a_sxl += xv * lg;
    a_sx  += xv;
    a_sgl += lgammaf(xv + 1.f);
  }
  __shared__ float sm[256];
  float r;
  r = block_reduce_sum(a_se,  sm); if (threadIdx.x == 0) atomicAdd(&se[b],  r);
  r = block_reduce_sum(a_sxl, sm); if (threadIdx.x == 0) atomicAdd(&sxl[b], r);
  r = block_reduce_sum(a_sx,  sm); if (threadIdx.x == 0) atomicAdd(&sx[b],  r);
  r = block_reduce_sum(a_sgl, sm); if (threadIdx.x == 0) atomicAdd(&sgl[b], r);
}

// ---- M = I + sD * WtW * sD / var ----
__global__ void vae_buildM(const float* WtW, const float* vlv, const float* lss, float* Mm) {
  int idx = blockIdx.x * blockDim.x + threadIdx.x;
  if (idx < 16384) {
    int i = idx >> 7, j = idx & 127;
    float var = expf(lss[0]);
    float sdi = expf(0.5f * vlv[i]);
    float sdj = expf(0.5f * vlv[j]);
    Mm[idx] = ((i == j) ? 1.f : 0.f) + sdi * WtW[idx] * sdj / var;
  }
}

// ---- Cholesky (128x128, SPD) + forward solve + final scalar ----
__global__ void vae_chol_final(const float* u, const float* vlv, const float* lss,
                               float* Mm, float* Yg,
                               const float* diff2, const float* se, const float* sxl,
                               const float* sx, const float* sgl, const float* prior,
                               const int* mxi, float* out, int dm1) {
  const int n = 128;
  int tid = threadIdx.x;  // 128 threads; thread owns row tid / batch tid
  __shared__ float red[128];

  // in-place Cholesky (lower triangle) in global scratch
  for (int k = 0; k < n; ++k) {
    if (tid == 0) Mm[k * n + k] = sqrtf(Mm[k * n + k]);
    __threadfence(); __syncthreads();
    float lkk = Mm[k * n + k];
    if (tid > k) Mm[tid * n + k] /= lkk;
    __threadfence(); __syncthreads();
    if (tid > k) {
      float lik = Mm[tid * n + k];
      for (int j = k + 1; j <= tid; ++j)
        Mm[tid * n + j] -= lik * Mm[j * n + k];
    }
    __threadfence(); __syncthreads();
  }

  // logdet(M) = 2 * sum log L_kk
  red[tid] = 2.f * logf(Mm[tid * n + tid]);
  __syncthreads();
  for (int s = 64; s; s >>= 1) {
    if (tid < s) red[tid] += red[tid + s];
    __syncthreads();
  }
  float logdetM = red[0];
  __syncthreads();

  // forward solve L w = u[b]*sD ; uMu = ||w||^2  (u^T M^-1 u)
  float lss0 = lss[0];
  float var  = expf(lss0);
  float uMu = 0.f;
  for (int i = 0; i < n; ++i) {
    float s = u[tid * n + i] * expf(0.5f * vlv[i]);
    for (int j = 0; j < i; ++j) s -= Mm[i * n + j] * Yg[j * n + tid];
    s /= Mm[i * n + i];
    Yg[i * n + tid] = s;
    uMu += s * s;
  }

  // per-batch terms
  int b = tid;
  float quad    = diff2[b] / var - uMu / (var * var);
  float logit_t = -0.5f * ((float)dm1 * LOG2PI + (float)dm1 * lss0 + logdetM + quad);
  int ik = mxi[b];
  float mx = __int_as_float(ik >= 0 ? ik : (ik ^ 0x7FFFFFFF));
  float mult_t = lgammaf(sx[b] + 1.f) - sgl[b] + sxl[b] - sx[b] * (mx + logf(se[b]));

  red[tid] = logit_t + mult_t;
  __syncthreads();
  for (int s = 64; s; s >>= 1) {
    if (tid < s) red[tid] += red[tid + s];
    __syncthreads();
  }
  if (tid == 0) {
    float mean_terms = red[0] / 128.f;
    float prior_loss = -0.5f * prior[0] / 16384.f - 0.5f * LOG2PI;
    out[0] = -(mean_terms + prior_loss);
  }
}

extern "C" void kernel_launch(void* const* d_in, const int* in_sizes, int n_in,
                              void* d_out, int out_size, void* d_ws, size_t ws_size,
                              hipStream_t stream) {
  const float* x     = (const float*)d_in[0];
  const int*   prow  = (const int*)d_in[1];
  const int*   pcol  = (const int*)d_in[2];
  const float* pval  = (const float*)d_in[3];
  const float* enc_w = (const float*)d_in[4];
  const float* dec_w = (const float*)d_in[5];
  const float* vlv   = (const float*)d_in[6];
  const float* lss   = (const float*)d_in[7];
  const float* eta   = (const float*)d_in[8];
  float* out = (float*)d_out;

  const int B = 128;
  const int D = in_sizes[0] / B;
  const int dm1 = D - 1;
  const int nnz = in_sizes[1];

  // workspace layout (floats)
  float* ws     = (float*)d_ws;
  float* lx     = ws;                          // B*D  (reused as diff after hx)
  float* diff   = lx;
  float* logits = lx + (size_t)B * D;          // B*D
  float* hx     = logits + (size_t)B * D;      // B*dm1
  float* zmean  = hx + (size_t)B * dm1;        // 128*128
  float* WtW    = zmean + 16384;               // 128*128
  float* uu     = WtW + 16384;                 // 128*128
  float* Mm     = uu + 16384;                  // 128*128
  float* Yg     = Mm + 16384;                  // 128*128
  float* diff2  = Yg + 16384;                  // 128
  float* se_    = diff2 + 128;                 // 128
  float* sxl_   = se_ + 128;                   // 128
  float* sx_    = sxl_ + 128;                  // 128
  float* sgl_   = sx_ + 128;                   // 128
  float* prior_ = sgl_ + 128;                  // 1
  int*   mxi    = (int*)(prior_ + 1);          // 128
  int*   rstart = mxi + 128;                   // dm1+1

  const int smallN = 5 * 16384 + 5 * 128 + 1;  // zmean..prior_
  vae_init_small<<<(smallN + 255) / 256, 256, 0, stream>>>(zmean, mxi, smallN);
  vae_precompute<<<(B * D + 255) / 256, 256, 0, stream>>>(x, lx, logits, B * D);
  vae_rowstart<<<(nnz + 255) / 256, 256, 0, stream>>>(prow, rstart, nnz, dm1);
  vae_hx<<<dim3(dm1, 16), 256, 0, stream>>>(lx, rstart, pcol, pval, hx, D, dm1);

  const int kch = 16;
  const int ky  = ((dm1 + 31) / 32 + kch - 1) / kch;
  // z_mean = hx(128 x dm1) @ enc_w^T : A(m,k)=hx[m*dm1+k], B(k,n)=enc_w[n*dm1+k]
  vae_gemm128<<<dim3(64, ky), 32, 0, stream>>>(hx, dm1, 1, enc_w, 1, dm1, zmean, dm1, kch);
  vae_prior<<<64, 256, 0, stream>>>(zmean, prior_);
  // WtW = dec_w^T dec_w : A(m,k)=dec_w[k*128+m], B(k,n)=dec_w[k*128+n]
  vae_gemm128<<<dim3(64, ky), 32, 0, stream>>>(dec_w, 1, 128, dec_w, 128, 1, WtW, dm1, kch);
  // mu + diff + per-batch sum diff^2 (diff overwrites lx region)
  vae_mu_diff<<<dim3((dm1 + 15) / 16, 8), 32, 0, stream>>>(zmean, dec_w, eta, diff, diff2, dm1);
  // u = diff @ dec_w : A(m,k)=diff[m*dm1+k], B(k,n)=dec_w[k*128+n]
  vae_gemm128<<<dim3(64, ky), 32, 0, stream>>>(diff, dm1, 1, dec_w, 128, 1, uu, dm1, kch);

  // multinomial pieces
  vae_scatter<<<dim3((nnz + 255) / 256, B), 256, 0, stream>>>(prow, pcol, pval, eta,
                                                              logits, D, dm1, nnz);
  vae_rowmax<<<dim3(64, B), 256, 0, stream>>>(logits, mxi, D);
  vae_rowsum<<<dim3(64, B), 256, 0, stream>>>(logits, x, mxi, se_, sxl_, sx_, sgl_, D);

  // Woodbury + final scalar
  vae_buildM<<<64, 256, 0, stream>>>(WtW, vlv, lss, Mm);
  vae_chol_final<<<1, 128, 0, stream>>>(uu, vlv, lss, Mm, Yg, diff2, se_, sxl_, sx_,
                                        sgl_, prior_, mxi, out, dm1);
}
